// HyperGCNBlock_51196010168985
// MI455X (gfx1250) — 16-bit and 8-bit matrix operand order repaired and checked by reading, compiled, not run on hardware
//
#include <hip/hip_runtime.h>
#include <hip/hip_bf16.h>

#define N_NODES 8192
#define E_HE    2048
#define NNZV    65536
#define D_FEAT  768
#define M_EDGES 262144
#define B_GRAPH 16
#define K_KEEP  6554
#define K_PAD   6656
#define MBLK_ROWS 32

typedef __attribute__((ext_vector_type(16))) _Float16 v16h;
typedef __attribute__((ext_vector_type(8)))  float    v8f;
typedef __attribute__((ext_vector_type(8)))  int      v8i;

#if defined(__has_builtin)
#if __has_builtin(__builtin_amdgcn_global_load_async_to_lds_b128)
#define HAVE_ASYNC_LDS 1
#endif
#endif

typedef int v4i_vs __attribute__((vector_size(16)));
typedef __attribute__((address_space(1))) v4i_vs as1_v4i_t;
typedef __attribute__((address_space(3))) v4i_vs as3_v4i_t;

__global__ void fill_u32_kernel(unsigned int* p, unsigned int v, size_t n) {
    size_t i = (size_t)blockIdx.x * blockDim.x + threadIdx.x;
    size_t stride = (size_t)gridDim.x * blockDim.x;
    for (; i < n; i += stride) p[i] = v;
}

__global__ __launch_bounds__(256) void gemm_xw_wmma(const float* __restrict__ x,
                                                    const float* __restrict__ W,
                                                    float* __restrict__ xw) {
    const int lane = threadIdx.x & 31;
    const int wave = threadIdx.x >> 5;
    const int tm   = blockIdx.x;
    const int tn   = blockIdx.y * 8 + wave;
    const int grp  = lane >> 4;
    const int l15  = lane & 15;
    const int arow = tm * 16 + l15;
    const int bcol = tn * 16 + l15;

    v8f c = {};
    for (int kt = 0; kt < D_FEAT / 32; ++kt) {
        const int kb = kt * 32;
        v16h ahi, alo, bhi, blo;
        const float* xa = x + (size_t)arow * D_FEAT + kb + 16 * grp;
        #pragma unroll
        for (int j = 0; j < 8; ++j) {
            float v0 = xa[j];
            float v1 = xa[j + 8];
            _Float16 h0 = (_Float16)v0;
            _Float16 h1 = (_Float16)v1;
            ahi[j]     = h0;  alo[j]     = (_Float16)(v0 - (float)h0);
            ahi[j + 8] = h1;  alo[j + 8] = (_Float16)(v1 - (float)h1);
        }
        const float* wb = W + (size_t)(kb + 16 * grp) * D_FEAT + bcol;
        #pragma unroll
        for (int h = 0; h < 16; ++h) {
            float v = wb[(size_t)h * D_FEAT];
            _Float16 hh = (_Float16)v;
            bhi[h] = hh;
            blo[h] = (_Float16)(v - (float)hh);
        }
        c = __builtin_amdgcn_wmma_f32_16x16x32_f16(false, ahi, false, bhi, (short)0, c, false, false);
        c = __builtin_amdgcn_wmma_f32_16x16x32_f16(false, ahi, false, blo, (short)0, c, false, false);
        c = __builtin_amdgcn_wmma_f32_16x16x32_f16(false, alo, false, bhi, (short)0, c, false, false);
    }
    #pragma unroll
    for (int r = 0; r < 8; ++r) {
        int m = tm * 16 + r + 8 * grp;
        xw[(size_t)m * D_FEAT + bcol] = c[r];
    }
}

__global__ void incidence_counts(const int* __restrict__ he, float* b_e, float* d_v) {
    int j = blockIdx.x * blockDim.x + threadIdx.x;
    if (j < NNZV) {
        atomicAdd(&d_v[he[j]], 1.0f);
        atomicAdd(&b_e[he[NNZV + j]], 1.0f);
    }
}

__global__ __launch_bounds__(256) void edge_accum(const int* __restrict__ he,
                                                  const float* __restrict__ xw,
                                                  float* __restrict__ ef) {
    int j = blockIdx.x;
    int n = he[j], e = he[NNZV + j];
    const float* src = xw + (size_t)n * D_FEAT;
    float* dst = ef + (size_t)e * D_FEAT;
    for (int d = threadIdx.x; d < D_FEAT; d += 256) atomicAdd(&dst[d], src[d]);
}

__global__ __launch_bounds__(256) void node_gather(const int* __restrict__ he,
                                                   const float* __restrict__ ef,
                                                   const float* __restrict__ b_e,
                                                   float* __restrict__ hg) {
    int j = blockIdx.x;
    int n = he[j], e = he[NNZV + j];
    float inv = 1.0f / fmaxf(b_e[e], 1.0f);
    const float* src = ef + (size_t)e * D_FEAT;
    float* dst = hg + (size_t)n * D_FEAT;
    for (int d = threadIdx.x; d < D_FEAT; d += 256) atomicAdd(&dst[d], src[d] * inv);
}

__global__ void wsum_kernel(const int* __restrict__ jei, const float* __restrict__ jw,
                            float* wsum) {
    int j = blockIdx.x * blockDim.x + threadIdx.x;
    if (j < M_EDGES) atomicAdd(&wsum[jei[M_EDGES + j]], jw[j]);
}

__global__ __launch_bounds__(256) void jac_agg(const int* __restrict__ jei,
                                               const float* __restrict__ jw,
                                               const float* __restrict__ xw,
                                               float* __restrict__ agg) {
    int j = blockIdx.x;
    int s = jei[j], t = jei[M_EDGES + j];
    float w = jw[j];
    const float* srow = xw + (size_t)s * D_FEAT;
    float* arow = agg + (size_t)t * D_FEAT;
    __builtin_prefetch(srow + threadIdx.x, 0, 1);
    for (int d = threadIdx.x; d < D_FEAT; d += 256) atomicAdd(&arow[d], w * srow[d]);
}

__global__ void cnt_kernel(const int* __restrict__ batch, float* cnt) {
    int i = blockIdx.x * blockDim.x + threadIdx.x;
    if (i < N_NODES) atomicAdd(&cnt[batch[i]], 1.0f);
}

__global__ __launch_bounds__(256) void z_kernel(float* __restrict__ hg,
                                                const float* __restrict__ agg,
                                                const float* __restrict__ d_v,
                                                const float* __restrict__ wsum,
                                                const float* __restrict__ bias,
                                                const int* __restrict__ batch,
                                                float* __restrict__ zg) {
    int i = blockIdx.x;
    float invd = 1.0f / fmaxf(d_v[i], 1.0f);
    float invw = 1.0f / fmaxf(wsum[i], 1e-12f);
    float* row = hg + (size_t)i * D_FEAT;
    const float* arow = agg + (size_t)i * D_FEAT;
    float* zgr = zg + (size_t)batch[i] * D_FEAT;
    for (int d = threadIdx.x; d < D_FEAT; d += 256) {
        float v = row[d] * invd + arow[d] * invw + bias[d];
        float g = 0.5f * v * (1.0f + erff(v * 0.70710678118654752f));
        row[d] = g;
        atomicAdd(&zgr[d], g);
    }
}

__global__ void zg_final(const float* __restrict__ zg, const float* __restrict__ cnt,
                         float* out) {
    int i = blockIdx.x * blockDim.x + threadIdx.x;
    if (i < B_GRAPH * D_FEAT) out[i] = zg[i] / fmaxf(cnt[i / D_FEAT], 1.0f);
}

__global__ void pnorm_kernel(const float* __restrict__ p, float* pn) {
    __shared__ float s[256];
    float acc = 0.f;
    for (int d = threadIdx.x; d < D_FEAT; d += 256) acc += p[d] * p[d];
    s[threadIdx.x] = acc; __syncthreads();
    for (int o = 128; o > 0; o >>= 1) { if (threadIdx.x < o) s[threadIdx.x] += s[threadIdx.x + o]; __syncthreads(); }
    if (threadIdx.x == 0) *pn = sqrtf(s[0]);
}

__global__ __launch_bounds__(256) void score_kernel(const float* __restrict__ z,
                                                    const float* __restrict__ p,
                                                    const float* __restrict__ pn,
                                                    float* score) {
    __shared__ float s[256];
    int i = blockIdx.x;
    const float* row = z + (size_t)i * D_FEAT;
    float acc = 0.f;
    for (int d = threadIdx.x; d < D_FEAT; d += 256) acc += row[d] * p[d];
    s[threadIdx.x] = acc; __syncthreads();
    for (int o = 128; o > 0; o >>= 1) { if (threadIdx.x < o) s[threadIdx.x] += s[threadIdx.x + o]; __syncthreads(); }
    if (threadIdx.x == 0) score[i] = s[0] / (*pn);
}

__global__ __launch_bounds__(256) void rank_kernel(const float* __restrict__ score, int* perm) {
    __shared__ int s[256];
    int i = blockIdx.x;
    float si = score[i];
    int cnt = 0;
    for (int j = threadIdx.x; j < N_NODES; j += 256) {
        float sj = score[j];
        cnt += (sj > si) || (sj == si && j < i);
    }
    s[threadIdx.x] = cnt; __syncthreads();
    for (int o = 128; o > 0; o >>= 1) { if (threadIdx.x < o) s[threadIdx.x] += s[threadIdx.x + o]; __syncthreads(); }
    if (threadIdx.x == 0) { int r = s[0]; if (r < K_KEEP) perm[r] = i; }
}

__global__ void pool_meta(const int* __restrict__ perm, const float* __restrict__ score,
                          const int* __restrict__ batch,
                          float* out_perm, float* out_sp, float* out_batch, int* new_id) {
    int r = blockIdx.x * blockDim.x + threadIdx.x;
    if (r < K_KEEP) {
        int i = perm[r];
        float sp = tanhf(score[i]);
        out_perm[r]  = (float)i;
        out_sp[r]    = sp;
        out_batch[r] = (float)batch[i];
        new_id[i] = r;
    }
}

__global__ __launch_bounds__(256) void z_pool_kernel(const int* __restrict__ perm,
                                                     const float* __restrict__ sp,
                                                     const float* __restrict__ z,
                                                     float* __restrict__ out_zp) {
    int r = blockIdx.x;
    int i = perm[r];
    float s = sp[r];
    const float* row = z + (size_t)i * D_FEAT;
    float* o = out_zp + (size_t)r * D_FEAT;
    for (int d = threadIdx.x; d < D_FEAT; d += 256) o[d] = row[d] * s;
}

__global__ void incidence_map(const int* __restrict__ he, const int* __restrict__ new_id,
                              float* out_ei, unsigned char* __restrict__ H) {
    int j = blockIdx.x * blockDim.x + threadIdx.x;
    if (j < NNZV) {
        int n = he[j], e = he[NNZV + j];
        int m = new_id[n];
        out_ei[j]        = (float)m;
        out_ei[NNZV + j] = (float)e;
        if (m >= 0) H[(size_t)m * E_HE + e] = 1;
    }
}

__global__ __launch_bounds__(256) void deg_kernel(const unsigned char* __restrict__ H, int* deg) {
    __shared__ int s[256];
    int r = blockIdx.x;
    const unsigned int* row = (const unsigned int*)(H + (size_t)r * E_HE);
    int acc = 0;
    for (int t = threadIdx.x; t < E_HE / 4; t += 256) {
        unsigned int u = row[t];
        acc += (int)((u & 0xFF) + ((u >> 8) & 0xFF) + ((u >> 16) & 0xFF) + ((u >> 24) & 0xFF));
    }
    s[threadIdx.x] = acc; __syncthreads();
    for (int o = 128; o > 0; o >>= 1) { if (threadIdx.x < o) s[threadIdx.x] += s[threadIdx.x + o]; __syncthreads(); }
    if (threadIdx.x == 0) deg[r] = s[0];
}

__global__ __launch_bounds__(256) void jaccard_wmma(const unsigned char* __restrict__ H,
                                                    const int* __restrict__ deg,
                                                    float* __restrict__ out_jac) {
    __shared__ unsigned char slab[MBLK_ROWS * E_HE];

    const int tid  = threadIdx.x;
    const unsigned char* gsrc = H + (size_t)(blockIdx.x * MBLK_ROWS) * E_HE;

#ifdef HAVE_ASYNC_LDS
    #pragma unroll
    for (int i = 0; i < 16; ++i) {
        int c = tid + 256 * i;
        __builtin_amdgcn_global_load_async_to_lds_b128(
            (as1_v4i_t*)(uintptr_t)(gsrc + (size_t)c * 16),
            (as3_v4i_t*)(uintptr_t)(unsigned)(uintptr_t)(slab + (size_t)c * 16),
            0, 0);
    }
#if __has_builtin(__builtin_amdgcn_s_wait_asynccnt)
    __builtin_amdgcn_s_wait_asynccnt(0);
#else
    asm volatile("s_wait_asynccnt 0" ::: "memory");
#endif
#else
    {
        const uint4* gs = (const uint4*)gsrc;
        uint4* ls = (uint4*)slab;
        #pragma unroll
        for (int i = 0; i < 16; ++i) { int c = tid + 256 * i; ls[c] = gs[c]; }
    }
#endif
    __syncthreads();

    const int lane = tid & 31;
    const int wave = tid >> 5;
    const int grp  = lane >> 4;
    const int l15  = lane & 15;
    const int tn0  = blockIdx.y * 16 + wave * 2;

    const unsigned char* brow0 = H + (size_t)(tn0 * 16 + l15) * E_HE;
    const unsigned char* brow1 = brow0 + (size_t)16 * E_HE;
    const unsigned char* arow0 = slab + (size_t)l15 * E_HE;
    const unsigned char* arow1 = slab + (size_t)(16 + l15) * E_HE;

    v8i acc00 = {}, acc01 = {}, acc10 = {}, acc11 = {};
    for (int kt = 0; kt < E_HE / 64; ++kt) {
        const int kbase = kt * 64;
        v8i a0, a1, b0, b1;
        #pragma unroll
        for (int v = 0; v < 8; ++v) {
            int ka = kbase + 32 * (v >> 2) + 16 * ((v >> 1) & 1) + 8 * grp + 4 * (v & 1);
            int kb = kbase + 32 * (v >> 2) + 16 * ((v >> 1) & 1) + 8 * grp + 4 * (v & 1);
            a0[v] = *(const int*)(arow0 + ka);
            a1[v] = *(const int*)(arow1 + ka);
            b0[v] = *(const int*)(brow0 + kb);
            b1[v] = *(const int*)(brow1 + kb);
        }
        acc00 = __builtin_amdgcn_wmma_i32_16x16x64_iu8(false, a0, false, b0, acc00, false, false);
        acc01 = __builtin_amdgcn_wmma_i32_16x16x64_iu8(false, a0, false, b1, acc01, false, false);
        acc10 = __builtin_amdgcn_wmma_i32_16x16x64_iu8(false, a1, false, b0, acc10, false, false);
        acc11 = __builtin_amdgcn_wmma_i32_16x16x64_iu8(false, a1, false, b1, acc11, false, false);
    }

    #pragma unroll
    for (int mt = 0; mt < 2; ++mt) {
        #pragma unroll
        for (int nt = 0; nt < 2; ++nt) {
            const v8i& acc = (mt == 0) ? (nt == 0 ? acc00 : acc01)
                                       : (nt == 0 ? acc10 : acc11);
            #pragma unroll
            for (int r = 0; r < 8; ++r) {
                int m = blockIdx.x * MBLK_ROWS + mt * 16 + r + 8 * grp;
                int n = (tn0 + nt) * 16 + l15;
                if (m < K_KEEP && n < K_KEEP) {
                    int inter = acc[r];
                    int un = deg[m] + deg[n] - inter;
                    float jac = (un > 0) ? (float)inter / (float)(un < 1 ? 1 : un) : 0.0f;
                    if (m == n) jac = 0.0f;
                    out_jac[(size_t)m * K_KEEP + n] = jac;
                }
            }
        }
    }
}

extern "C" void kernel_launch(void* const* d_in, const int* in_sizes, int n_in,
                              void* d_out, int out_size, void* d_ws, size_t ws_size,
                              hipStream_t stream) {
    (void)in_sizes; (void)n_in; (void)out_size; (void)ws_size;
    const float* x     = (const float*)d_in[0];
    const int*   he    = (const int*)d_in[1];
    const int*   jei   = (const int*)d_in[2];
    const float* jw    = (const float*)d_in[3];
    const int*   batch = (const int*)d_in[4];
    const float* W     = (const float*)d_in[5];
    const float* bias  = (const float*)d_in[6];
    const float* p     = (const float*)d_in[7];

    float* xw   = (float*)d_ws;
    float* ef   = xw   + (size_t)N_NODES * D_FEAT;
    float* hg   = ef   + (size_t)E_HE   * D_FEAT;
    float* agg  = hg   + (size_t)N_NODES * D_FEAT;
    float* b_e  = agg  + (size_t)N_NODES * D_FEAT;
    float* d_v  = b_e  + E_HE;
    float* wsum = d_v  + N_NODES;
    float* zg   = wsum + N_NODES;
    float* cnt  = zg   + (size_t)B_GRAPH * D_FEAT;
    float* pn   = cnt  + B_GRAPH;
    float* score = pn + 1;
    int*   perm  = (int*)(score + N_NODES);
    int*   newid = perm + K_KEEP;
    int*   deg   = newid + N_NODES;
    unsigned char* H = (unsigned char*)(((uintptr_t)(deg + K_PAD) + 255) & ~(uintptr_t)255);

    float* out       = (float*)d_out;
    const size_t SZ_ZP = (size_t)K_KEEP * D_FEAT;
    float* out_zp    = out;
    float* out_ei    = out_zp + SZ_ZP;
    float* out_jac   = out_ei + 2 * (size_t)NNZV;
    float* out_batch = out_jac + (size_t)K_KEEP * K_KEEP;
    float* out_zg    = out_batch + K_KEEP;
    float* out_perm  = out_zg + (size_t)B_GRAPH * D_FEAT;
    float* out_sp    = out_perm + K_KEEP;

    const size_t span_words = (size_t)E_HE * D_FEAT + 2 * (size_t)N_NODES * D_FEAT
                            + E_HE + 2 * (size_t)N_NODES + (size_t)B_GRAPH * D_FEAT + B_GRAPH;
    fill_u32_kernel<<<4096, 256, 0, stream>>>((unsigned int*)ef, 0u, span_words);
    fill_u32_kernel<<<4096, 256, 0, stream>>>((unsigned int*)H, 0u, (size_t)K_PAD * E_HE / 4);
    fill_u32_kernel<<<256, 256, 0, stream>>>((unsigned int*)newid, 0xFFFFFFFFu, N_NODES);

    gemm_xw_wmma<<<dim3(N_NODES / 16, (D_FEAT / 16) / 8), 256, 0, stream>>>(x, W, xw);

    incidence_counts<<<NNZV / 256, 256, 0, stream>>>(he, b_e, d_v);
    edge_accum<<<NNZV, 256, 0, stream>>>(he, xw, ef);
    node_gather<<<NNZV, 256, 0, stream>>>(he, ef, b_e, hg);

    wsum_kernel<<<M_EDGES / 256, 256, 0, stream>>>(jei, jw, wsum);
    jac_agg<<<M_EDGES, 256, 0, stream>>>(jei, jw, xw, agg);

    cnt_kernel<<<N_NODES / 256, 256, 0, stream>>>(batch, cnt);
    z_kernel<<<N_NODES, 256, 0, stream>>>(hg, agg, d_v, wsum, bias, batch, zg);
    zg_final<<<(B_GRAPH * D_FEAT + 255) / 256, 256, 0, stream>>>(zg, cnt, out_zg);

    pnorm_kernel<<<1, 256, 0, stream>>>(p, pn);
    score_kernel<<<N_NODES, 256, 0, stream>>>(hg, p, pn, score);
    rank_kernel<<<N_NODES, 256, 0, stream>>>(score, perm);
    pool_meta<<<(K_KEEP + 255) / 256, 256, 0, stream>>>(perm, score, batch,
                                                        out_perm, out_sp, out_batch, newid);
    z_pool_kernel<<<K_KEEP, 256, 0, stream>>>(perm, out_sp, hg, out_zp);

    incidence_map<<<NNZV / 256, 256, 0, stream>>>(he, newid, out_ei, H);
    deg_kernel<<<K_KEEP, 256, 0, stream>>>(H, deg);

    jaccard_wmma<<<dim3(K_PAD / MBLK_ROWS, K_PAD / 256), 256, 0, stream>>>(H, deg, out_jac);
}
